// MultiHeadAttention_24893630448214
// MI455X (gfx1250) — matrix operand order repaired and checked by reading, not compiled, not run on hardware
//
#include <hip/hip_runtime.h>

// ---------------------------------------------------------------------------
// MI455X (gfx1250, wave32) fused multi-head attention block.
// Matmuls via v_wmma_f32_16x16x32_f16; tile staging via async global->LDS
// (global_load_async_to_lds_b128 + s_wait_asynccnt), the CDNA5 DMA-style path.
// ---------------------------------------------------------------------------

typedef __attribute__((ext_vector_type(16))) _Float16 v16h;
typedef __attribute__((ext_vector_type(8)))  float    v8f;

union AFrag { v16h h; uint32_t u[8]; };
union H2    { uint32_t u; _Float16 h[2]; };

__device__ __forceinline__ v8f wmma_f16(v16h a, v16h b, v8f c) {
  // (neg_a, A, neg_b, B, c_mod, C, reuse_a, reuse_b)
  return __builtin_amdgcn_wmma_f32_16x16x32_f16(false, a, false, b,
                                                (short)0, c, false, false);
}

// Async global->LDS copy, 16 bytes per lane. lds_off = byte offset within the
// workgroup's LDS allocation (low 32 bits of a generic LDS pointer).
__device__ __forceinline__ void async_b128(uint32_t lds_off, const void* g) {
  asm volatile("global_load_async_to_lds_b128 %0, %1, off"
               :: "v"(lds_off), "v"(g)
               : "memory");
}

__device__ __forceinline__ void wait_async0() {
#if __has_builtin(__builtin_amdgcn_s_wait_asynccnt)
  __builtin_amdgcn_s_wait_asynccnt(0);
#else
  asm volatile("s_wait_asynccnt 0x0" ::: "memory");
#endif
}

#define TT 2048
#define HH 16
#define HDIM 64
#define CC 1024

// ---------------------------------------------------------------------------
// fp32 -> f16 convert
// ---------------------------------------------------------------------------
__global__ void f32_to_f16(const float* __restrict__ src,
                           _Float16* __restrict__ dst, int n) {
  int i = blockIdx.x * blockDim.x + threadIdx.x;
  if (i < n) dst[i] = (_Float16)src[i];
}

// ---------------------------------------------------------------------------
// WMMA GEMM: C[M,N] f32 = A[M,K] f16 (row major) * B[K,N] f16 (row major)
// Block = 128 threads (4 waves). Block tile 128x64, wave tile 32x64,
// K-steps of 32. A tile via async-to-LDS b128; B tile transposed to [n][k].
// Requires M%128==0, N%64==0, K%32==0.
// ---------------------------------------------------------------------------
__global__ __launch_bounds__(128) void gemm_f16_f32(
    const _Float16* __restrict__ A, const _Float16* __restrict__ B,
    float* __restrict__ C, int M, int N, int K)
{
  // Row strides are multiples of 16B so async b128 stores stay aligned.
  __shared__ __align__(16) _Float16 As[128][40];  // [m][k], 80B stride
  __shared__ __align__(16) _Float16 Bs[64][34];   // [n][k]

  const int tid  = threadIdx.x;
  const int lane = tid & 31;
  const int wave = tid >> 5;
  const int tm   = blockIdx.y * 128;
  const int tn   = blockIdx.x * 64;
  const int m16  = lane & 15;
  const int hi   = lane >> 4;           // 0: lanes 0-15, 1: lanes 16-31

  v8f acc[2][4] = {};

  for (int k0 = 0; k0 < K; k0 += 32) {
    // Stage A 128x32 halves = 512 x 16B chunks via async DMA (4 per thread)
    #pragma unroll
    for (int i = 0; i < 4; ++i) {
      int idx = tid + i * 128;          // 0..511
      int r = idx >> 2, ch = idx & 3;
      async_b128((uint32_t)(size_t)&As[r][ch * 8],
                 &A[(size_t)(tm + r) * K + k0 + ch * 8]);
    }
    // Stage B 32x64 transposed into [n][k] (dword loads, paired half stores)
    for (int i = tid; i < 32 * 32; i += 128) {
      int kk = i >> 5, np = (i & 31) << 1;
      H2 cv;
      cv.u = *(const uint32_t*)&B[(size_t)(k0 + kk) * N + tn + np];
      Bs[np][kk]     = cv.h[0];
      Bs[np + 1][kk] = cv.h[1];
    }
    if (k0 + 32 < K)   // gfx1250 global_prefetch_b8 of next B K-slab
      __builtin_prefetch(&B[(size_t)(k0 + 32 + (tid >> 2)) * N + tn + (tid & 3) * 16], 0, 0);
    wait_async0();
    __syncthreads();

    // A fragments: 16x32, ISA layout (lanes<16: K0-7,16-23; lanes>=16: +8)
    AFrag af[2];
    #pragma unroll
    for (int rb = 0; rb < 2; ++rb)
      #pragma unroll
      for (int v = 0; v < 8; ++v) {
        int kk = (v < 4) ? (hi * 8 + 2 * v) : (16 + hi * 8 + 2 * (v - 4));
        af[rb].u[v] = *(const uint32_t*)&As[wave * 32 + rb * 16 + m16][kk];
      }
    #pragma unroll
    for (int sub = 0; sub < 4; ++sub) {
      AFrag bf;   // B fragment [n][k]: lanes<16 K0-7,16-23; lanes>=16: +8
      #pragma unroll
      for (int v = 0; v < 8; ++v)
        bf.u[v] = *(const uint32_t*)&Bs[sub * 16 + m16][(v < 4) ? (hi * 8 + 2 * v) : (16 + hi * 8 + 2 * (v - 4))];
      #pragma unroll
      for (int rb = 0; rb < 2; ++rb)
        acc[rb][sub] = wmma_f16(af[rb].h, bf.h, acc[rb][sub]);
    }
    __syncthreads();
  }

  // C/D layout: VGPR r -> row r (lanes 0-15) / row r+8 (lanes 16-31)
  #pragma unroll
  for (int rb = 0; rb < 2; ++rb) {
    const int rbase = tm + wave * 32 + rb * 16 + hi * 8;
    const int cbase = tn + m16;
    #pragma unroll
    for (int sub = 0; sub < 4; ++sub)
      #pragma unroll
      for (int r = 0; r < 8; ++r)
        C[(size_t)(rbase + r) * N + cbase + sub * 16] = acc[rb][sub][r];
  }
}

// ---------------------------------------------------------------------------
// RoPE + split. qkvf: [B,T,3C] f32. Writes f16 q/k/v in [B,H,T,hd] layout and
// fp32 k_t / v_t outputs. One thread per (b,t,h,d2), d2 in [0,32).
// ---------------------------------------------------------------------------
__global__ void rope_split(const float* __restrict__ qkvf,
                           _Float16* __restrict__ qh, _Float16* __restrict__ kh,
                           _Float16* __restrict__ vh,
                           float* __restrict__ kt, float* __restrict__ vt)
{
  int idx = blockIdx.x * blockDim.x + threadIdx.x;   // 2*2048*16*32 = 2^21
  int d2 = idx & 31;
  int h  = (idx >> 5) & 15;
  int t  = (idx >> 9) & 2047;
  int b  = idx >> 20;

  const float* base = qkvf + ((size_t)(b * TT + t)) * 3072;
  float q1 = base[h * 64 + 2 * d2],        q2 = base[h * 64 + 2 * d2 + 1];
  float k1 = base[1024 + h * 64 + 2 * d2], k2 = base[1024 + h * 64 + 2 * d2 + 1];
  float v1 = base[2048 + h * 64 + d2],     v2 = base[2048 + h * 64 + d2 + 32];

  // inv_freq = 10000^(-2*d2/64) = exp(-d2 * ln(10000)/32)
  float invf = __expf(-0.28782313662f * (float)d2);
  float ang  = (float)t * invf;
  float s, c;
  __sincosf(ang, &s, &c);

  float qa = q1 * c - q2 * s, qb = q1 * s + q2 * c;
  float ka = k1 * c - k2 * s, kb = k1 * s + k2 * c;

  size_t o = ((size_t)((b * HH + h) * TT) + t) * HDIM;
  qh[o + d2] = (_Float16)qa;  qh[o + d2 + 32] = (_Float16)qb;
  kh[o + d2] = (_Float16)ka;  kh[o + d2 + 32] = (_Float16)kb;
  vh[o + d2] = (_Float16)v1;  vh[o + d2 + 32] = (_Float16)v2;
  kt[o + d2] = ka;            kt[o + d2 + 32] = kb;
  vt[o + d2] = v1;            vt[o + d2 + 32] = v2;
}

// ---------------------------------------------------------------------------
// Flash attention. Grid: (T/64, B*H). Block = 128 threads = 4 waves.
// Each wave owns 16 query rows. K tile staged via async-to-LDS; V tile
// transposed via VALU. S=Q*K^T and acc+=P*V via WMMA; online softmax with
// wave32 shuffle reductions. Causal mask only on the diagonal tile.
// ---------------------------------------------------------------------------
__global__ __launch_bounds__(128) void attn_fused(
    const _Float16* __restrict__ qh, const _Float16* __restrict__ kh,
    const _Float16* __restrict__ vh, _Float16* __restrict__ aout)
{
  __shared__ __align__(16) _Float16 Ks[64][72];   // [key][dim], 144B stride
  __shared__ _Float16 Vs[64][66];                 // [dim][key] (transposed)
  __shared__ _Float16 Ps[4][16][66];              // per-wave P staging

  const int bh = blockIdx.y;
  const int b  = bh >> 4, h = bh & 15;
  const int qtile = blockIdx.x;
  const int qbase = qtile * 64;
  const int tid  = threadIdx.x;
  const int lane = tid & 31, wave = tid >> 5;
  const int m16  = lane & 15, hi = lane >> 4;

  // Load this wave's Q fragments once (16 rows x 64 dims -> 2 A-frags)
  const _Float16* Qb = qh + ((size_t)bh * TT + qbase + wave * 16) * HDIM;
  AFrag qf[2];
  #pragma unroll
  for (int kb = 0; kb < 2; ++kb)
    #pragma unroll
    for (int v = 0; v < 8; ++v) {
      int kk = ((v < 4) ? (hi * 8 + 2 * v) : (16 + hi * 8 + 2 * (v - 4))) + kb * 32;
      qf[kb].u[v] = *(const uint32_t*)&Qb[(size_t)m16 * HDIM + kk];
    }

  float mrow[8], lrow[8];
  v8f acc[4] = {};
  #pragma unroll
  for (int r = 0; r < 8; ++r) { mrow[r] = -1e30f; lrow[r] = 0.f; }

  const float scale = 0.125f;  // 1/sqrt(64)
  const int rowbase = qbase + wave * 16 + hi * 8;

  for (int j = 0; j <= qtile; ++j) {
    const _Float16* Kb = kh + ((size_t)bh * TT + j * 64) * HDIM;
    const _Float16* Vb = vh + ((size_t)bh * TT + j * 64) * HDIM;

    // Stage K 64x64 halves = 512 x 16B chunks via async DMA (4 per thread)
    #pragma unroll
    for (int i = 0; i < 4; ++i) {
      int idx = tid + i * 128;          // 0..511
      int key = idx >> 3, ch = idx & 7;
      async_b128((uint32_t)(size_t)&Ks[key][ch * 8],
                 &Kb[(size_t)key * HDIM + ch * 8]);
    }
    // Stage V transposed [dim][key]
    for (int i = tid; i < 64 * 32; i += 128) {
      int key = i >> 5, c = (i & 31) << 1;
      H2 cv;
      cv.u = *(const uint32_t*)&Vb[(size_t)key * HDIM + c];
      Vs[c][key]     = cv.h[0];
      Vs[c + 1][key] = cv.h[1];
    }
    wait_async0();
    __syncthreads();

    // S = Q * K^T  (K in [n=key][k=dim] layout == Ks)
    v8f s[4];
    #pragma unroll
    for (int sub = 0; sub < 4; ++sub) {
      v8f cfr = {};
      #pragma unroll
      for (int kb = 0; kb < 2; ++kb) {
        AFrag bf;
        #pragma unroll
        for (int v = 0; v < 8; ++v)
          bf.u[v] = *(const uint32_t*)&Ks[sub * 16 + m16][((v < 4) ? (hi * 8 + 2 * v) : (16 + hi * 8 + 2 * (v - 4))) + kb * 32];
        cfr = wmma_f16(qf[kb].h, bf.h, cfr);
      }
      s[sub] = cfr;
    }

    // Online softmax (per-row stats; 16-lane shuffle reductions)
    const bool diag = (j == qtile);
    const int colbase = j * 64;
    float p[4][8];
    #pragma unroll
    for (int r = 0; r < 8; ++r) {
      float mx = mrow[r];
      const int row = rowbase + r;
      #pragma unroll
      for (int sub = 0; sub < 4; ++sub) {
        float sv = s[sub][r] * scale;
        if (diag && (colbase + sub * 16 + m16) > row) sv = -1e30f;
        p[sub][r] = sv;
        mx = fmaxf(mx, sv);
      }
      #pragma unroll
      for (int off = 1; off < 16; off <<= 1)
        mx = fmaxf(mx, __shfl_xor(mx, off, 32));
      float alpha = __expf(mrow[r] - mx);
      float rs = 0.f;
      #pragma unroll
      for (int sub = 0; sub < 4; ++sub) {
        float e = __expf(p[sub][r] - mx);
        p[sub][r] = e;
        rs += e;
      }
      #pragma unroll
      for (int off = 1; off < 16; off <<= 1)
        rs += __shfl_xor(rs, off, 32);
      lrow[r] = lrow[r] * alpha + rs;
      mrow[r] = mx;
      #pragma unroll
      for (int sub = 0; sub < 4; ++sub)
        acc[sub][r] *= alpha;
    }

    // Bounce P through per-wave LDS: C-layout -> A-fragment layout
    #pragma unroll
    for (int sub = 0; sub < 4; ++sub)
      #pragma unroll
      for (int r = 0; r < 8; ++r)
        Ps[wave][hi * 8 + r][sub * 16 + m16] = (_Float16)p[sub][r];

    AFrag pf[2];
    #pragma unroll
    for (int kb = 0; kb < 2; ++kb)
      #pragma unroll
      for (int v = 0; v < 8; ++v) {
        int kk = ((v < 4) ? (hi * 8 + 2 * v) : (16 + hi * 8 + 2 * (v - 4))) + kb * 32;
        pf[kb].u[v] = *(const uint32_t*)&Ps[wave][m16][kk];
      }

    // acc += P * V  (V in [n=dim][k=key] layout == Vs)
    #pragma unroll
    for (int dn = 0; dn < 4; ++dn)
      #pragma unroll
      for (int kb = 0; kb < 2; ++kb) {
        AFrag bf;
        #pragma unroll
        for (int v = 0; v < 8; ++v)
          bf.u[v] = *(const uint32_t*)&Vs[dn * 16 + m16][((v < 4) ? (hi * 8 + 2 * v) : (16 + hi * 8 + 2 * (v - 4))) + kb * 32];
        acc[dn] = wmma_f16(pf[kb].h, bf.h, acc[dn]);
      }
    __syncthreads();
  }

  // Normalize, write f16 [B,T,C] for the output projection
  #pragma unroll
  for (int dn = 0; dn < 4; ++dn)
    #pragma unroll
    for (int r = 0; r < 8; ++r) {
      int row = rowbase + r;
      int d   = dn * 16 + m16;
      float o = acc[dn][r] / lrow[r];
      aout[((size_t)b * TT + row) * CC + h * HDIM + d] = (_Float16)o;
    }
}

// ---------------------------------------------------------------------------
// Launch
// ---------------------------------------------------------------------------
extern "C" void kernel_launch(void* const* d_in, const int* in_sizes, int n_in,
                              void* d_out, int out_size, void* d_ws, size_t ws_size,
                              hipStream_t stream)
{
  const float* x     = (const float*)d_in[0];
  // d_in[1] = attn_mask (causal tril) -- computed analytically, unused
  const float* w_qkv = (const float*)d_in[2];
  const float* w_out = (const float*)d_in[3];
  float* out = (float*)d_out;

  const size_t BTC  = (size_t)2 * TT * CC;        // 4,194,304
  const size_t WQKV = (size_t)CC * 3 * CC;        // 3,145,728
  const size_t WOUT = (size_t)CC * CC;            // 1,048,576
  const size_t QKV  = (size_t)2 * TT * 3 * CC;    // 12,582,912

  char* w = (char*)d_ws;
  _Float16* xh    = (_Float16*)w; w += BTC  * 2;
  _Float16* wqkvh = (_Float16*)w; w += WQKV * 2;
  _Float16* wouth = (_Float16*)w; w += WOUT * 2;
  float*    qkvf  = (float*)w;    w += QKV  * 4;
  _Float16* qh    = (_Float16*)w; w += BTC * 2;
  _Float16* kh    = (_Float16*)w; w += BTC * 2;
  _Float16* vh    = (_Float16*)w; w += BTC * 2;
  _Float16* ah    = (_Float16*)w; w += BTC * 2;

  f32_to_f16<<<(unsigned)((BTC  + 255) / 256), 256, 0, stream>>>(x, xh, (int)BTC);
  f32_to_f16<<<(unsigned)((WQKV + 255) / 256), 256, 0, stream>>>(w_qkv, wqkvh, (int)WQKV);
  f32_to_f16<<<(unsigned)((WOUT + 255) / 256), 256, 0, stream>>>(w_out, wouth, (int)WOUT);

  // qkv = x @ w_qkv : M=4096, N=3072, K=1024
  gemm_f16_f32<<<dim3(3072 / 64, 4096 / 128), 128, 0, stream>>>(
      xh, wqkvh, qkvf, 4096, 3072, 1024);

  float* kt = out + BTC;
  float* vt = out + 2 * BTC;
  rope_split<<<(2u * TT * HH * 32) / 256, 256, 0, stream>>>(qkvf, qh, kh, vh, kt, vt);

  attn_fused<<<dim3(TT / 64, 2 * HH), 128, 0, stream>>>(qh, kh, vh, ah);

  // out = attn @ w_out : M=4096, N=1024, K=1024
  gemm_f16_f32<<<dim3(1024 / 64, 4096 / 128), 128, 0, stream>>>(
      ah, wouth, out, 4096, 1024, 1024);
}
